// Recurrence_31705448579081
// MI455X (gfx1250) — compile-verified, matrix operand order checked by reading, not run on hardware
//
#include <hip/hip_runtime.h>
#include <math.h>

// ---------------- problem constants ----------------
#define TT   256
#define NN   512
#define LL   64
#define HH   256
#define NAA  16
#define OC   276          // output cols per (t,n): a,b,probs(16),v,h(256),p
#define INC  67           // input cols per (t,n)

typedef __bf16 bf16;
typedef bf16  v16bf __attribute__((ext_vector_type(16)));
typedef float v8f   __attribute__((ext_vector_type(8)));

// ---------------- WMMA helpers ----------------
__device__ __forceinline__ v8f wmma_bf16(v16bf a, v16bf b, v8f c) {
    return __builtin_amdgcn_wmma_f32_16x16x32_bf16(false, a, false, b, (short)0, c,
                                                   false, false);
}

// A fragment: 16 rows x 32 K from LDS row-major [16][stride]
__device__ __forceinline__ v16bf load_a(const bf16* A, int stride, int kbase, int lane) {
    return *(const v16bf*)(A + (lane & 15) * stride + kbase + ((lane >> 4) << 4));
}

// B fragment: packed weight tiles, 512 bf16 per (ntile,ktile)
__device__ __forceinline__ v16bf load_b(const bf16* Wp, int tileIdx, int lane) {
    return *(const v16bf*)(Wp + ((size_t)tileIdx << 9) + (lane << 4));
}

__device__ __forceinline__ float sigmoidf_(float x) { return 1.f / (1.f + expf(-x)); }

// store C tile (+bias, optional relu) as bf16 into LDS [16][256]
__device__ __forceinline__ void store_c_bf16(bf16* dst, int nbase, int lane, v8f c,
                                             const float* bias, bool relu) {
    int col = nbase + (lane & 15);
    float b = bias[col];
    int rb = (lane >> 4) << 3;
#pragma unroll
    for (int j = 0; j < 8; ++j) {
        float v = c[j] + b;
        if (relu && v < 0.f) v = 0.f;
        dst[(rb + j) * HH + col] = (bf16)v;
    }
}

// one MLP layer: [16 x 256] = A[16 x K] @ Wp^T, K = nKt*32; 2 column-tiles per wave
__device__ __forceinline__ void mlp_layer(const bf16* A, int astride, const bf16* Wp,
                                          int nKt, bf16* dst, const float* bias,
                                          bool relu, int wave, int lane) {
#pragma unroll
    for (int tt = 0; tt < 2; ++tt) {
        int nt = wave * 2 + tt;
        v8f acc = {};
        for (int kt = 0; kt < nKt; ++kt) {
            v16bf a = load_a(A, astride, kt * 32, lane);
            v16bf b = load_b(Wp, nt * nKt + kt, lane);
            acc = wmma_bf16(a, b, acc);
        }
        store_c_bf16(dst, nt * 16, lane, acc, bias, relu);
    }
}

// ---------------- kernel: pack fp32 weight (Nout x K) into bf16 WMMA B tiles ----------------
__global__ void pack_weight_kernel(const float* __restrict__ W, bf16* __restrict__ Wp,
                                   int Nout, int K) {
    int total = Nout * K;
    int nKt = K >> 5;
    for (int idx = blockIdx.x * blockDim.x + threadIdx.x; idx < total;
         idx += gridDim.x * blockDim.x) {
        int n = idx / K, k = idx - n * K;
        int nt = n >> 4, nr = n & 15, kt = k >> 5, kk = k & 31;
        int lane = ((kk >> 4) << 4) | nr;
        int i = kk & 15;
        Wp[(((size_t)(nt * nKt + kt)) << 9) + lane * 16 + i] = (bf16)W[idx];
    }
}

// ---------------- kernel: P recurrence + trivial outputs (a, b, p) ----------------
__global__ void p_kernel(const float* __restrict__ inputs, const float* __restrict__ hxs,
                         int* __restrict__ Pws, float* __restrict__ out) {
    int n = blockIdx.x * blockDim.x + threadIdx.x;
    if (n >= NN) return;
    int P = (int)hxs[(size_t)n * OC + 275];
    for (int t = 0; t < TT; ++t) {
        Pws[t * NN + n] = P;
        int a = (int)inputs[((size_t)t * NN + n) * INC + 66];
        int b = (a == NAA - 1) ? 1 : 0;
        P = (P + b) & (LL - 1);
        size_t ob = ((size_t)t * NN + n) * OC;
        out[ob + 0]   = (float)a;
        out[ob + 1]   = (float)b;
        out[ob + 275] = (float)P;
        if (t == TT - 1) {
            size_t tb = ((size_t)TT * NN + n) * OC;
            out[tb + 0] = (float)a; out[tb + 1] = (float)b; out[tb + 275] = (float)P;
        }
    }
}

// ---------------- kernel: broadcast constant h into output (incl. tail) ----------------
__global__ void hbroadcast_kernel(const float* __restrict__ hxs, float* __restrict__ out) {
    const size_t total = (size_t)(TT + 1) * NN * HH;
    size_t stride = (size_t)gridDim.x * blockDim.x;
    for (size_t idx = (size_t)blockIdx.x * blockDim.x + threadIdx.x; idx < total;
         idx += stride) {
        size_t tn = idx / HH;
        int j = (int)(idx - tn * HH);
        int n = (int)(tn % NN);
        out[tn * OC + 19 + j] = hxs[(size_t)n * OC + 19 + j];
    }
}

// ---------------- kernel: bidirectional GRU over L=64 ----------------
// grid: dir(2) x batch-tile(8) = 16 blocks, 256 threads (8 waves)
__global__ __launch_bounds__(256) void gru_kernel(const float* __restrict__ inputs,
                                                  const float* __restrict__ wih,
                                                  const float* __restrict__ bih,
                                                  const float* __restrict__ bhh,
                                                  const bf16* __restrict__ whhp0,
                                                  const bf16* __restrict__ whhp1,
                                                  bf16* __restrict__ Mb) {
    __shared__ __align__(64) bf16 sH[64 * HH];   // 32 KB, batch-tile hidden state
    int dir = blockIdx.x >> 3;
    int n0 = (blockIdx.x & 7) * 64;
    int tid = threadIdx.x, wave = tid >> 5, lane = tid & 31;
    const bf16*  Wp   = dir ? whhp1 : whhp0;
    const float* wihd = wih + dir * 768;   // gru_wih (2,768,1)
    const float* bihd = bih + dir * 768;
    const float* bhhd = bhh + dir * 768;

    for (int i = tid; i < 64 * HH; i += 256) sH[i] = (bf16)0.f;
    __syncthreads();

    for (int s = 0; s < LL; ++s) {
        int l = dir ? (LL - 1 - s) : s;
        for (int mt = 0; mt < 4; ++mt) {
            v8f acc[3][2] = {};
            const bf16* Abase = sH + mt * 16 * HH;
            for (int kt = 0; kt < 8; ++kt) {
                v16bf a = load_a(Abase, HH, kt * 32, lane);
#pragma unroll
                for (int g = 0; g < 3; ++g)
#pragma unroll
                    for (int ts = 0; ts < 2; ++ts) {
                        int ntile = g * 16 + wave * 2 + ts;
                        v16bf b = load_b(Wp, ntile * 8 + kt, lane);
                        acc[g][ts] = wmma_bf16(a, b, acc[g][ts]);
                    }
            }
            __syncthreads();   // all waves finished reading rows of this M-tile
#pragma unroll
            for (int ts = 0; ts < 2; ++ts) {
                int col = wave * 32 + ts * 16 + (lane & 15);
#pragma unroll
                for (int j = 0; j < 8; ++j) {
                    int row = mt * 16 + ((lane >> 4) << 3) + j;
                    int n = n0 + row;
                    float x  = inputs[(size_t)n * INC + l];         // sizes[0][n][l]
                    float gr = x * wihd[col]       + bihd[col]       + acc[0][ts][j] + bhhd[col];
                    float gz = x * wihd[256 + col] + bihd[256 + col] + acc[1][ts][j] + bhhd[256 + col];
                    float inn = x * wihd[512 + col] + bihd[512 + col];
                    float hn  = acc[2][ts][j] + bhhd[512 + col];
                    float rg = sigmoidf_(gr);
                    float zg = sigmoidf_(gz);
                    float ng = tanhf(inn + rg * hn);
                    float hold = (float)sH[row * HH + col];
                    float hnew = (1.f - zg) * ng + zg * hold;
                    sH[row * HH + col] = (bf16)hnew;
                    Mb[((size_t)l * NN + n) * 512 + dir * 256 + col] = (bf16)hnew;
                }
            }
            __syncthreads();
        }
    }
}

// ---------------- kernel: gh = h0 @ ctrl_whh.T + ctrl_bhh  (512x768, K=256) ----------------
__global__ __launch_bounds__(256) void gh_kernel(const float* __restrict__ hxs,
                                                 const float* __restrict__ bhh,
                                                 const bf16* __restrict__ Wp,
                                                 float* __restrict__ gh) {
    int wid = blockIdx.x * 8 + (threadIdx.x >> 5);
    int lane = threadIdx.x & 31;
    int mt = wid / 48, nt = wid % 48;
    if (mt >= 32) return;
    const float* hrow = hxs + ((size_t)(mt * 16 + (lane & 15))) * OC + 19;
    int khalf = (lane >> 4) << 4;
    v8f acc = {};
    for (int kt = 0; kt < 8; ++kt) {
        v16bf a;
#pragma unroll
        for (int i = 0; i < 16; ++i) a[i] = (bf16)hrow[kt * 32 + khalf + i];
        v16bf b = load_b(Wp, nt * 8 + kt, lane);
        acc = wmma_bf16(a, b, acc);
    }
    int col = nt * 16 + (lane & 15);
    int rb = (lane >> 4) << 3;
#pragma unroll
    for (int j = 0; j < 8; ++j)
        gh[(size_t)(mt * 16 + rb + j) * 768 + col] = acc[j] + bhh[col];
}

// ---------------- kernel: fused controller, one block per (t, 16-row n-tile) ----------------
__global__ __launch_bounds__(256) void ctrl_kernel(
    const float* __restrict__ inputs, const float* __restrict__ hxs,
    const float* __restrict__ embA, const float* __restrict__ embP,
    const float* __restrict__ embI,
    const bf16* __restrict__ wihp, const float* __restrict__ cbih,
    const float* __restrict__ gh,
    const bf16* __restrict__ aw0p, const float* __restrict__ ab0,
    const bf16* __restrict__ aw1p, const float* __restrict__ ab1,
    const float* __restrict__ aw2, const float* __restrict__ ab2,
    const bf16* __restrict__ cw0p, const float* __restrict__ cb0,
    const bf16* __restrict__ cw1p, const float* __restrict__ cb1,
    const float* __restrict__ cw2, const float* __restrict__ cb2,
    const bf16* __restrict__ bw0p, const float* __restrict__ bb0,
    const bf16* __restrict__ bw1p, const float* __restrict__ bb1,
    const float* __restrict__ bw2, const float* __restrict__ bb2,
    const int* __restrict__ Pws, const bf16* __restrict__ Mb,
    float* __restrict__ out) {

    __shared__ __align__(64) bf16 sEmb[16 * 768];  // 24 KB; reused as t1/t2 after GEMM1
    __shared__ __align__(64) bf16 sR[16 * 512];    // 16 KB gathered r (kept for actor)
    __shared__ __align__(64) bf16 sY[16 * HH];     // 8 KB controller output y
    __shared__ float sB[16];                       // beta sigmoid per row
    __shared__ float sLg[16 * 16];                 // actor logits

    int t  = blockIdx.x >> 5;
    int n0 = (blockIdx.x & 31) << 4;
    int tid = threadIdx.x, wave = tid >> 5, lane = tid & 31;

    // ---- stage 0: gather embeddings + r into LDS (bf16) ----
    {
        int m = tid >> 4, part = tid & 15;
        int n = n0 + m;
        const float* inrow = inputs + ((size_t)t * NN + n) * INC;
        int idx_t = (int)inrow[65];
        int pos_t = (int)inrow[64];
        int aprev = (t > 0) ? (int)inputs[((size_t)(t - 1) * NN + n) * INC + 66]
                            : (int)hxs[(size_t)n * OC + 0];
        const float* e0 = embI + (size_t)idx_t * HH;
        const float* e1 = embP + (size_t)pos_t * HH;
        const float* e2 = embA + (size_t)aprev * HH;
#pragma unroll
        for (int i = 0; i < 16; ++i) {
            int c = part * 16 + i;
            sEmb[m * 768 + c]       = (bf16)e0[c];
            sEmb[m * 768 + 256 + c] = (bf16)e1[c];
            sEmb[m * 768 + 512 + c] = (bf16)e2[c];
        }
        int P = Pws[t * NN + n];
        const bf16* mrow = Mb + ((size_t)P * NN + n) * 512;
#pragma unroll
        for (int i = 0; i < 32; ++i) sR[m * 512 + part * 32 + i] = mrow[part * 32 + i];
    }
    __syncthreads();

    // ---- stage 1: gi = X @ ctrl_wih.T (16x1280x768) + fused GRU-cell gates → y ----
    {
        v8f acc[3][2] = {};
        for (int kt = 0; kt < 40; ++kt) {
            v16bf a = (kt < 24) ? load_a(sEmb, 768, kt * 32, lane)
                                : load_a(sR, 512, kt * 32 - 768, lane);
#pragma unroll
            for (int g = 0; g < 3; ++g)
#pragma unroll
                for (int ts = 0; ts < 2; ++ts) {
                    int ntile = g * 16 + wave * 2 + ts;
                    v16bf b = load_b(wihp, ntile * 40 + kt, lane);
                    acc[g][ts] = wmma_bf16(a, b, acc[g][ts]);
                }
        }
#pragma unroll
        for (int ts = 0; ts < 2; ++ts) {
            int col = wave * 32 + ts * 16 + (lane & 15);
#pragma unroll
            for (int j = 0; j < 8; ++j) {
                int m = ((lane >> 4) << 3) + j;
                int n = n0 + m;
                const float* ghrow = gh + (size_t)n * 768;
                float gr  = acc[0][ts][j] + cbih[col]       + ghrow[col];
                float gz  = acc[1][ts][j] + cbih[256 + col] + ghrow[256 + col];
                float inn = acc[2][ts][j] + cbih[512 + col];
                float hn  = ghrow[512 + col];
                float rg = sigmoidf_(gr);
                float zg = sigmoidf_(gz);
                float ng = tanhf(inn + rg * hn);
                float h0 = hxs[(size_t)n * OC + 19 + col];
                float y  = (1.f - zg) * ng + zg * h0;
                sY[m * HH + col] = (bf16)y;
            }
        }
    }
    __syncthreads();

    bf16* t1 = sEmb;             // reuse embedding region as MLP scratch
    bf16* t2 = sEmb + 16 * HH;

    // ---- beta MLP on y ----
    mlp_layer(sY, HH, bw0p, 8, t1, bb0, false, wave, lane);
    __syncthreads();
    mlp_layer(t1, HH, bw1p, 8, t2, bb1, true, wave, lane);
    __syncthreads();
    // beta final dot (+sigmoid) concurrently with critic L0
    if (tid < 16) {
        float s = bb2[0];
        for (int k = 0; k < HH; ++k) s += (float)t2[tid * HH + k] * bw2[k];
        sB[tid] = sigmoidf_(s);
    }
    mlp_layer(sY, HH, cw0p, 8, t1, cb0, false, wave, lane);
    __syncthreads();
    mlp_layer(t1, HH, cw1p, 8, t2, cb1, true, wave, lane);
    __syncthreads();
    // critic final dot concurrently with actor L0 (reads sR)
    if (tid < 16) {
        float s = cb2[0];
        for (int k = 0; k < HH; ++k) s += (float)t2[tid * HH + k] * cw2[k];
        size_t ob = ((size_t)t * NN + n0 + tid) * OC;
        out[ob + 18] = s;
        if (t == TT - 1) out[((size_t)TT * NN + n0 + tid) * OC + 18] = s;
    }
    mlp_layer(sR, 512, aw0p, 16, t1, ab0, false, wave, lane);
    __syncthreads();
    mlp_layer(t1, HH, aw1p, 8, t2, ab1, true, wave, lane);
    __syncthreads();
    // actor logits (15 per row)
    if (tid < 240) {
        int m = tid / 15, j = tid - m * 15;
        float s = ab2[j];
        for (int k = 0; k < HH; ++k) s += (float)t2[m * HH + k] * aw2[j * HH + k];
        sLg[m * 16 + j] = s;
    }
    __syncthreads();
    // softmax + mixture with beta, write probs
    if (tid < 16) {
        float b = sB[tid];
        float mx = -1e30f;
        for (int j = 0; j < 15; ++j) mx = fmaxf(mx, sLg[tid * 16 + j]);
        float sum = 0.f;
        float e[15];
        for (int j = 0; j < 15; ++j) { e[j] = expf(sLg[tid * 16 + j] - mx); sum += e[j]; }
        float inv = (1.f - b) / sum;
        size_t ob = ((size_t)t * NN + n0 + tid) * OC;
        size_t tb = ((size_t)TT * NN + n0 + tid) * OC;
        for (int j = 0; j < 15; ++j) {
            float p = e[j] * inv;
            out[ob + 2 + j] = p;
            if (t == TT - 1) out[tb + 2 + j] = p;
        }
        out[ob + 17] = b;
        if (t == TT - 1) out[tb + 17] = b;
    }
}

// ---------------- host launcher ----------------
extern "C" void kernel_launch(void* const* d_in, const int* in_sizes, int n_in,
                              void* d_out, int out_size, void* d_ws, size_t ws_size,
                              hipStream_t stream) {
    (void)in_sizes; (void)n_in; (void)out_size; (void)ws_size;

    const float* inputs  = (const float*)d_in[0];
    const float* hxs     = (const float*)d_in[1];
    const float* embA    = (const float*)d_in[2];
    const float* embP    = (const float*)d_in[3];
    const float* embI    = (const float*)d_in[4];
    const float* gru_wih = (const float*)d_in[5];
    const float* gru_whh = (const float*)d_in[6];
    const float* gru_bih = (const float*)d_in[7];
    const float* gru_bhh = (const float*)d_in[8];
    const float* c_wih   = (const float*)d_in[9];
    const float* c_whh   = (const float*)d_in[10];
    const float* c_bih   = (const float*)d_in[11];
    const float* c_bhh   = (const float*)d_in[12];
    const float* a_w0 = (const float*)d_in[13]; const float* a_b0 = (const float*)d_in[14];
    const float* a_w1 = (const float*)d_in[15]; const float* a_b1 = (const float*)d_in[16];
    const float* a_w2 = (const float*)d_in[17]; const float* a_b2 = (const float*)d_in[18];
    const float* k_w0 = (const float*)d_in[19]; const float* k_b0 = (const float*)d_in[20];
    const float* k_w1 = (const float*)d_in[21]; const float* k_b1 = (const float*)d_in[22];
    const float* k_w2 = (const float*)d_in[23]; const float* k_b2 = (const float*)d_in[24];
    const float* b_w0 = (const float*)d_in[25]; const float* b_b0 = (const float*)d_in[26];
    const float* b_w1 = (const float*)d_in[27]; const float* b_b1 = (const float*)d_in[28];
    const float* b_w2 = (const float*)d_in[29]; const float* b_b2 = (const float*)d_in[30];
    float* out = (float*)d_out;

    // workspace layout
    char* ws = (char*)d_ws;
    int*   Pws   = (int*)ws;                                   // 256*512 ints
    float* ghbuf = (float*)(ws + 524288);                      // 512*768 f32
    bf16*  Mb    = (bf16*)(ws + 2097152);                      // 64*512*512 bf16
    bf16*  pk    = (bf16*)(ws + 35651584);
    bf16* whhp0 = pk;                 // 768x256
    bf16* whhp1 = pk + 196608;        // 768x256
    bf16* cwhhp = pk + 393216;        // 768x256
    bf16* cwihp = pk + 589824;        // 768x1280
    bf16* aw0p  = pk + 1572864;       // 256x512
    bf16* aw1p  = pk + 1703936;       // 256x256
    bf16* bw0p  = pk + 1769472;
    bf16* bw1p  = pk + 1835008;
    bf16* cw0p  = pk + 1900544;
    bf16* cw1p  = pk + 1966080;

    auto pack = [&](const float* W, bf16* Wp, int Nout, int K) {
        int total = Nout * K;
        pack_weight_kernel<<<(total + 255) / 256, 256, 0, stream>>>(W, Wp, Nout, K);
    };
    pack(gru_whh,            whhp0, 768, 256);
    pack(gru_whh + 196608,   whhp1, 768, 256);
    pack(c_whh,              cwhhp, 768, 256);
    pack(c_wih,              cwihp, 768, 1280);
    pack(a_w0,               aw0p,  256, 512);
    pack(a_w1,               aw1p,  256, 256);
    pack(b_w0,               bw0p,  256, 256);
    pack(b_w1,               bw1p,  256, 256);
    pack(k_w0,               cw0p,  256, 256);
    pack(k_w1,               cw1p,  256, 256);

    p_kernel<<<2, 256, 0, stream>>>(inputs, hxs, Pws, out);
    hbroadcast_kernel<<<4096, 256, 0, stream>>>(hxs, out);
    gru_kernel<<<16, 256, 0, stream>>>(inputs, gru_wih, gru_bih, gru_bhh,
                                       whhp0, whhp1, Mb);
    gh_kernel<<<192, 256, 0, stream>>>(hxs, c_bhh, cwhhp, ghbuf);
    ctrl_kernel<<<TT * 32, 256, 0, stream>>>(
        inputs, hxs, embA, embP, embI, cwihp, c_bih, ghbuf,
        aw0p, a_b0, aw1p, a_b1, a_w2, a_b2,
        cw0p, k_b0, cw1p, k_b1, k_w2, k_b2,
        bw0p, b_b0, bw1p, b_b1, b_w2, b_b2,
        Pws, Mb, out);
}
